// DeformableConvBlock_A_16750372455126
// MI455X (gfx1250) — compile-verified, matrix operand order checked by reading, not run on hardware
//
#include <hip/hip_runtime.h>

typedef __attribute__((ext_vector_type(16))) _Float16 v16h;
typedef __attribute__((ext_vector_type(8)))  float    v8f;

#define HW_   4096
#define CIN   256
#define COUT  256
#define CI2   512
#define K2_   9
#define NOFF  18

// Element e of a v16h A/B fragment maps to K index within the 32-wide chunk:
// lanes 0-15 hold K = {0..7,16..23}, lanes 16-31 hold K = {8..15,24..31}.
__device__ __forceinline__ int kkmap(int e, int half) {
  return ((e < 8) ? e : (e + 8)) + half * 8;
}

__device__ __forceinline__ void f32_split(float v, _Float16& hi, _Float16& lo) {
  hi = (_Float16)v;
  lo = (_Float16)(v - (float)hi);
}

__device__ __forceinline__ v8f wmma3(v16h ah, v16h al, v16h bh, v16h bl, v8f c) {
  c = __builtin_amdgcn_wmma_f32_16x16x32_f16(false, ah, false, bh, (short)0, c, false, false);
  c = __builtin_amdgcn_wmma_f32_16x16x32_f16(false, ah, false, bl, (short)0, c, false, false);
  c = __builtin_amdgcn_wmma_f32_16x16x32_f16(false, al, false, bh, (short)0, c, false, false);
  return c;
}

// gfx1250 async global->LDS copy (ASYNCcnt). Builtins don't cover these; inline asm.
__device__ __forceinline__ void async_b128(unsigned int lds_byte_off, unsigned long long gaddr) {
  asm volatile("global_load_async_to_lds_b128 %0, %1, off"
               :: "v"(lds_byte_off), "v"(gaddr) : "memory");
}
__device__ __forceinline__ void wait_async_all() {
  asm volatile("s_wait_asynccnt 0x0" ::: "memory");
}

// ---------- pack dconv_w (256,256,3,3) -> fragment-ordered f16 hi/lo ----------
// K = k2*256 + c, 72 chunks of 32; 16 N-tiles. idx = ((q*16+t)*32+lane)*16+e
__global__ void pack_bd_kernel(const float* __restrict__ w,
                               _Float16* __restrict__ bhi, _Float16* __restrict__ blo) {
  int tid = blockIdx.x * blockDim.x + threadIdx.x;
  if (tid >= 72 * 16 * 32 * 16) return;
  int e = tid & 15;
  int lane = (tid >> 4) & 31;
  int tt = tid >> 9;
  int t = tt & 15;
  int q = tt >> 4;                       // 0..71
  int k2 = q >> 3;
  int c = ((q & 7) << 5) + kkmap(e, lane >> 4);
  int n = (t << 4) + (lane & 15);
  float v = w[(n * CIN + c) * K2_ + k2];
  _Float16 hi, lo; f32_split(v, hi, lo);
  bhi[tid] = hi; blo[tid] = lo;
}

// ---------- pack offs_w (18,512,3,3) -> fragments, N padded to 32 ----------
// K = k2*512 + c, 144 chunks of 32; 2 N-tiles.
__global__ void pack_bo_kernel(const float* __restrict__ w,
                               _Float16* __restrict__ bhi, _Float16* __restrict__ blo) {
  int tid = blockIdx.x * blockDim.x + threadIdx.x;
  if (tid >= 144 * 2 * 32 * 16) return;
  int e = tid & 15;
  int lane = (tid >> 4) & 31;
  int tt = tid >> 9;
  int t = tt & 1;
  int q = tt >> 1;                       // 0..143
  int k2 = q >> 4;
  int c = ((q & 15) << 5) + kkmap(e, lane >> 4);
  int n = (t << 4) + (lane & 15);
  float v = (n < NOFF) ? w[(n * CI2 + c) * K2_ + k2] : 0.0f;
  _Float16 hi, lo; f32_split(v, hi, lo);
  bhi[tid] = hi; blo[tid] = lo;
}

// ---------- offset conv: implicit GEMM M=16384,K=4608,N=32(pad) ----------
// Block = 8 waves; each wave: one 16-row M tile, both N tiles.
__global__ __launch_bounds__(256)
void offconv_wmma_kernel(const float* __restrict__ ci,
                         const v16h* __restrict__ Bh, const v16h* __restrict__ Bl,
                         const float* __restrict__ offs_b,
                         float* __restrict__ offset_out) {
  int lane = threadIdx.x & 31;
  int wave = threadIdx.x >> 5;
  int half = lane >> 4;
  int mrow = lane & 15;
  int mbase = blockIdx.x * 128 + wave * 16;
  int b   = mbase >> 12;
  int rem = mbase & 4095;
  int h   = rem >> 6;
  int w0  = rem & 63;
  int wpix = w0 + mrow;
  const float* cib = ci + (size_t)b * CI2 * HW_;

  v8f acc[2]; acc[0] = {}; acc[1] = {};

  for (int k2 = 0; k2 < 9; ++k2) {
    int ky = k2 / 3, kx = k2 % 3;
    int y = h - 1 + ky, x = wpix - 1 + kx;
    bool valid = (y >= 0) && (y < 64) && (x >= 0) && (x < 64);
    int base = y * 64 + x;
    for (int cb = 0; cb < 16; ++cb) {
      int q = k2 * 16 + cb;
      int c0 = cb << 5;
      v16h ah, al;
#pragma unroll
      for (int e = 0; e < 16; ++e) {
        int coff = (c0 + kkmap(e, half)) * HW_;
        float v = valid ? cib[coff + base] : 0.0f;
        _Float16 hi, lo; f32_split(v, hi, lo);
        ah[e] = hi; al[e] = lo;
      }
#pragma unroll
      for (int t = 0; t < 2; ++t) {
        v16h bh = Bh[(q * 2 + t) * 32 + lane];
        v16h bl = Bl[(q * 2 + t) * 32 + lane];
        acc[t] = wmma3(ah, al, bh, bl, acc[t]);
      }
    }
  }
  // D tile: VGPR r -> row M = r + 8*half, col N = lane%16
#pragma unroll
  for (int t = 0; t < 2; ++t) {
    int n = t * 16 + (lane & 15);
    if (n < NOFF) {
      float bias = offs_b[n];
#pragma unroll
      for (int r = 0; r < 8; ++r) {
        int wp = w0 + r + half * 8;
        offset_out[((size_t)b * NOFF + n) * HW_ + h * 64 + wp] = acc[t][r] + bias;
      }
    }
  }
}

// ---------- deformable conv: implicit GEMM M=16384,K=2304,N=256 ----------
// Block = 8 waves = 8 M tiles; each wave owns ALL 16 N tiles (16x256 per wave).
// B fragments double-buffer-staged block-wide in LDS via async global->LDS.
__global__ __launch_bounds__(256)
void deform_wmma_kernel(const float* __restrict__ pi,
                        const float* __restrict__ offset,
                        const _Float16* __restrict__ Bh, const _Float16* __restrict__ Bl,
                        const float* __restrict__ dconv_b,
                        float* __restrict__ out) {
  __shared__ _Float16 smem[2][2][8192];   // [buf][hi/lo][16KB of fragments]
  int tid  = threadIdx.x;
  int lane = tid & 31;
  int wave = tid >> 5;
  int half = lane >> 4;
  int mrow = lane & 15;
  int mbase = blockIdx.x * 128 + wave * 16;
  int b   = mbase >> 12;
  int rem = mbase & 4095;
  int h   = rem >> 6;
  int w0  = rem & 63;
  int wpix = w0 + mrow;
  const float* pib = pi + (size_t)b * CIN * HW_;

  auto stage = [&](int q, int buf) {
    const char* srcH = (const char*)Bh + (size_t)q * 16384;
    const char* srcL = (const char*)Bl + (size_t)q * 16384;
    unsigned int dH = (unsigned int)(uintptr_t)&smem[buf][0][0];
    unsigned int dL = (unsigned int)(uintptr_t)&smem[buf][1][0];
#pragma unroll
    for (int i = 0; i < 4; ++i) {
      int o = i * 4096 + tid * 16;
      async_b128(dH + o, (unsigned long long)(srcH + o));
      async_b128(dL + o, (unsigned long long)(srcL + o));
    }
  };

  v8f acc[16];
#pragma unroll
  for (int t = 0; t < 16; ++t) acc[t] = {};

  stage(0, 0);
  wait_async_all();
  __syncthreads();

  for (int k2 = 0; k2 < 9; ++k2) {
    // Per-(m,k2) bilinear metadata, constant across the 8 c-chunks of this tap.
    float dy = offset[((size_t)b * NOFF + 2 * k2    ) * HW_ + h * 64 + wpix];
    float dx = offset[((size_t)b * NOFF + 2 * k2 + 1) * HW_ + h * 64 + wpix];
    float sy = (float)(h - 1 + k2 / 3) + dy;
    float sx = (float)(wpix - 1 + k2 % 3) + dx;
    float fy0 = floorf(sy), fx0 = floorf(sx);
    int y0 = (int)fy0, x0 = (int)fx0;
    int y1 = y0 + 1,  x1 = x0 + 1;
    float wy1 = sy - fy0, wx1 = sx - fx0;
    float wy0 = 1.0f - wy1, wx0 = 1.0f - wx1;
    bool vy0 = (y0 >= 0) && (y0 < 64), vy1 = (y1 >= 0) && (y1 < 64);
    bool vx0 = (x0 >= 0) && (x0 < 64), vx1 = (x1 >= 0) && (x1 < 64);
    float w00 = (vy0 && vx0) ? wy0 * wx0 : 0.0f;
    float w01 = (vy0 && vx1) ? wy0 * wx1 : 0.0f;
    float w10 = (vy1 && vx0) ? wy1 * wx0 : 0.0f;
    float w11 = (vy1 && vx1) ? wy1 * wx1 : 0.0f;
    int y0c = min(max(y0, 0), 63), y1c = min(max(y1, 0), 63);
    int x0c = min(max(x0, 0), 63), x1c = min(max(x1, 0), 63);
    int a00 = y0c * 64 + x0c, a01 = y0c * 64 + x1c;
    int a10 = y1c * 64 + x0c, a11 = y1c * 64 + x1c;

    for (int cb = 0; cb < 8; ++cb) {
      int q = k2 * 8 + cb;
      int buf = q & 1;
      if (q + 1 < 72) stage(q + 1, buf ^ 1);   // prefetch next chunk's B

      int c0 = cb << 5;
      v16h ah, al;
#pragma unroll
      for (int e = 0; e < 16; ++e) {
        int coff = (c0 + kkmap(e, half)) * HW_;
        float v = w00 * pib[coff + a00] + w01 * pib[coff + a01]
                + w10 * pib[coff + a10] + w11 * pib[coff + a11];
        _Float16 hi, lo; f32_split(v, hi, lo);
        ah[e] = hi; al[e] = lo;
      }
      const v16h* sBh = (const v16h*)&smem[buf][0][0];
      const v16h* sBl = (const v16h*)&smem[buf][1][0];
#pragma unroll
      for (int t = 0; t < 16; ++t) {
        v16h bh = sBh[t * 32 + lane];
        v16h bl = sBl[t * 32 + lane];
        acc[t] = wmma3(ah, al, bh, bl, acc[t]);
      }
      wait_async_all();
      __syncthreads();
    }
  }
#pragma unroll
  for (int t = 0; t < 16; ++t) {
    int n = t * 16 + (lane & 15);
    float bias = dconv_b[n];
#pragma unroll
    for (int r = 0; r < 8; ++r) {
      out[((size_t)b * COUT + n) * HW_ + h * 64 + (w0 + r + half * 8)] = acc[t][r] + bias;
    }
  }
}

extern "C" void kernel_launch(void* const* d_in, const int* in_sizes, int n_in,
                              void* d_out, int out_size, void* d_ws, size_t ws_size,
                              hipStream_t stream) {
  const float* ci      = (const float*)d_in[0];
  const float* pi      = (const float*)d_in[1];
  const float* offs_w  = (const float*)d_in[2];
  const float* offs_b  = (const float*)d_in[3];
  const float* dconv_w = (const float*)d_in[4];
  const float* dconv_b = (const float*)d_in[5];
  float* out = (float*)d_out;

  char* ws = (char*)d_ws;
  float*    offbuf = (float*)ws;                       // 4*18*64*64*4   = 1,179,648 B
  _Float16* bd_hi  = (_Float16*)(ws + 1179648);        // 589824*2       = 1,179,648 B
  _Float16* bd_lo  = (_Float16*)(ws + 2359296);        //                  1,179,648 B
  _Float16* bo_hi  = (_Float16*)(ws + 3538944);        // 147456*2       =   294,912 B
  _Float16* bo_lo  = (_Float16*)(ws + 3833856);        //                    294,912 B
                                                       // total ~4.13 MB

  pack_bd_kernel<<<2304, 256, 0, stream>>>(dconv_w, bd_hi, bd_lo);
  pack_bo_kernel<<<576, 256, 0, stream>>>(offs_w, bo_hi, bo_lo);
  offconv_wmma_kernel<<<128, 256, 0, stream>>>(ci, (const v16h*)bo_hi, (const v16h*)bo_lo,
                                               offs_b, offbuf);
  deform_wmma_kernel<<<128, 256, 0, stream>>>(pi, offbuf, bd_hi, bd_lo,
                                              dconv_b, out);
}
